// K1CompleteSystem_40346922778765
// MI455X (gfx1250) — matrix operand order repaired and checked by reading, not compiled, not run on hardware
//
#include <hip/hip_runtime.h>
#include <hip/hip_bf16.h>
#include <math.h>

#define BDIM 512   // batch
#define DDIM 512   // model dim
#define HDIM 2048  // hidden dim
#define CNUM 4

typedef __attribute__((ext_vector_type(16))) _Float16 v16h;
typedef __attribute__((ext_vector_type(8)))  _Float16 v8h;
typedef __attribute__((ext_vector_type(4)))  _Float16 v4h;
typedef __attribute__((ext_vector_type(8)))  float    v8f;
typedef __attribute__((ext_vector_type(4)))  float    v4f;

#define EPI_GELU  0
#define EPI_RESID 1
#define EPI_STORE 2

#define LDS_PITCH 40   // halves per B row in LDS (pad 32->40: banks r*20 mod 64 distinct)

// ---------------- init: cur = x, routing state = 0 ----------------
__global__ void init_kernel(const float* __restrict__ x, float* __restrict__ cur,
                            int* __restrict__ idx, int n)
{
    int i = blockIdx.x * blockDim.x + threadIdx.x;
    if (i < n) cur[i] = x[i];
    if (i == 0) { idx[0] = 0; idx[1] = 0; }
}

// ---------------- gather + convert one agent's weight matrix to f16 ----------------
__global__ void convw_kernel(const float* __restrict__ W, _Float16* __restrict__ out,
                             const int* __restrict__ idxslot, int off, int elems)
{
    int z    = blockIdx.z;
    int gidx = off + idxslot[0] + z;
    const float* src = W + (size_t)gidx * elems;
    _Float16*    dst = out + (size_t)z * elems;
    int i = (blockIdx.x * 256 + threadIdx.x) * 4;
    if (i < elems) {
        v4f f = *(const v4f*)(src + i);
        v4h o;
        #pragma unroll
        for (int c = 0; c < 4; c++) o[c] = (_Float16)f[c];
        *(v4h*)(dst + i) = o;
    }
}

// ---------------- layernorm (f32 in -> f16 out), per-agent params ----------------
__global__ void ln_kernel(const float* __restrict__ in, _Float16* __restrict__ out,
                          const float* __restrict__ lw, const float* __restrict__ lb,
                          const int* __restrict__ idxslot, int off, int L,
                          size_t inSlotStride, size_t outSlotStride)
{
    int row = blockIdx.x;
    int y   = blockIdx.y;                 // agent slot (0 for parent, 0..3 for children)
    int gidx = off + idxslot[0] + y;
    const float* ip = in  + (size_t)y * inSlotStride  + (size_t)row * L;
    _Float16*    op = out + (size_t)y * outSlotStride + (size_t)row * L;
    const float* gw = lw + (size_t)gidx * L;
    const float* gb = lb + (size_t)gidx * L;

    float s = 0.f, s2 = 0.f;
    for (int i = threadIdx.x; i < L; i += 256) { float v = ip[i]; s += v; s2 += v * v; }
    __shared__ float sh[512];
    sh[threadIdx.x] = s; sh[256 + threadIdx.x] = s2;
    __syncthreads();
    for (int st = 128; st > 0; st >>= 1) {
        if (threadIdx.x < st) {
            sh[threadIdx.x]       += sh[threadIdx.x + st];
            sh[256 + threadIdx.x] += sh[256 + threadIdx.x + st];
        }
        __syncthreads();
    }
    float mean = sh[0]   / (float)L;
    float var  = sh[256] / (float)L - mean * mean;
    float rstd = rsqrtf(var + 1e-5f);
    for (int i = threadIdx.x; i < L; i += 256) {
        float v = (ip[i] - mean) * rstd * gw[i] + gb[i];
        op[i] = (_Float16)v;
    }
}

// ---------------- WMMA GEMM: Out[M,N] = A[M,K](f16) x W16[N,K](f16)^T ----------------
// block = 4 waves = 256M x 64N ; wave = 64M x 64N (16 accumulators)
// B tile (32K x 64N) double-buffered in LDS: one barrier per K-step,
// next tile's global->LDS stage overlaps current tile's 16 WMMAs
__global__ __launch_bounds__(128)
void wmma_gemm(const _Float16* __restrict__ A, const _Float16* __restrict__ W16,
               const float* __restrict__ bias, float* __restrict__ Out,
               const int* __restrict__ idxslot, int off,
               int M, int N, int K, size_t aSlotStride, size_t outSlotStride, int epi)
{
    int z    = blockIdx.z;
    int gidx = off + idxslot[0] + z;                 // bias only
    const _Float16* Ap = A   + (size_t)z * aSlotStride;
    const _Float16* Wp = W16 + (size_t)z * (size_t)N * (size_t)K;
    const float*    bp = bias + (size_t)gidx * N;
    float*          Op = Out + (size_t)z * outSlotStride;

    int tid  = threadIdx.x;
    int lane = tid & 31;
    int wv   = tid >> 5;                             // wave in block 0..3
    int nblk = N >> 6;
    int tn   = (blockIdx.x % nblk) << 6;
    int tm   = ((blockIdx.x / nblk) << 8) + (wv << 6);
    int lh = lane & 15, hi = lane >> 4;

    // B staging: thread t -> row sn = t/2 (0..63), k-part sk = (t&1)*16
    int sn = tid >> 1;
    int sk = (tid & 1) << 4;

    __shared__ _Float16 lb[2][64 * LDS_PITCH];       // 2 x 5 KB double buffer

    v8f acc[16] = {};

    // A layout (16-bit A 16x32): lanes 0-15 rows, K {koff..+7, koff+16..+23}, koff = hi*8
    const _Float16* arow   = Ap + (size_t)(tm + lh) * K + hi * 8;
    const _Float16* gstage = Wp + (size_t)(tn + sn) * K + sk;
    _Float16* sst0 = &lb[0][sn * LDS_PITCH + sk];
    _Float16* sst1 = &lb[1][sn * LDS_PITCH + sk];

    // prologue: stage tile kk=0 into buffer 0
    {
        v8h b0 = *(const v8h*)gstage;
        v8h b1 = *(const v8h*)(gstage + 8);
        *(v8h*)(sst0)     = b0;
        *(v8h*)(sst0 + 8) = b1;
    }

    int p = 0;
    for (int kk = 0; kk < K; kk += 32) {
        __syncthreads();                 // buf p ready for reads; buf p^1 free for writes
        if (kk + 32 < K) {               // stage next tile into the other buffer
            const _Float16* gp = gstage + kk + 32;
            v8h b0 = *(const v8h*)gp;
            v8h b1 = *(const v8h*)(gp + 8);
            _Float16* d = p ? sst0 : sst1;
            *(v8h*)(d)     = b0;
            *(v8h*)(d + 8) = b1;
            __builtin_prefetch(gp + 32, 0, 3);
        }
        // A fragments for the wave's 4 M-subtiles
        v16h af[4];
        #pragma unroll
        for (int mi = 0; mi < 4; mi++) {
            const _Float16* ap = arow + (size_t)(mi * 16) * K + kk;
            v8h a0 = *(const v8h*)ap;
            v8h a1 = *(const v8h*)(ap + 16);
            #pragma unroll
            for (int i = 0; i < 8; i++) { af[mi][i] = a0[i]; af[mi][8 + i] = a1[i]; }
        }
        // preload all 4 B fragments (one dscnt wait covers all, WMMAs run back-to-back)
        v16h bf[4];
        #pragma unroll
        for (int j = 0; j < 4; j++) {
            // B layout (16-bit B 32x16): lane = column n%16, K {koff..+7, koff+16..+23}, koff = hi*8
            const _Float16* lp = &lb[p][(j * 16 + lh) * LDS_PITCH + hi * 8];
            v8h b0 = *(const v8h*)lp;
            v8h b1 = *(const v8h*)(lp + 16);
            #pragma unroll
            for (int i = 0; i < 8; i++) { bf[j][i] = b0[i]; bf[j][8 + i] = b1[i]; }
        }
        #pragma unroll
        for (int j = 0; j < 4; j++) {
            #pragma unroll
            for (int mi = 0; mi < 4; mi++)
                acc[j * 4 + mi] = __builtin_amdgcn_wmma_f32_16x16x32_f16(
                    false, af[mi], false, bf[j], (short)0, acc[j * 4 + mi], false, false);
        }
        p ^= 1;
    }

    // C/D layout: VGPR r -> M = sub + r + 8*hi ; N = tn + j*16 + (lane%16)
    #pragma unroll
    for (int j = 0; j < 4; j++) {
        int col = tn + j * 16 + lh;
        float bv = bp[col];
        #pragma unroll
        for (int mi = 0; mi < 4; mi++) {
            #pragma unroll
            for (int r = 0; r < 8; r++) {
                int row = tm + mi * 16 + r + 8 * hi;
                float v = acc[j * 4 + mi][r] + bv;
                size_t o = (size_t)row * N + col;
                if (epi == EPI_GELU) {
                    v = 0.5f * v * (1.f + erff(v * 0.70710678118654752f));  // exact gelu
                    Op[o] = v;
                } else if (epi == EPI_RESID) {
                    Op[o] = 2.f * Op[o] + v;      // cur = cur + (mm + b2 + cur)
                } else {
                    Op[o] = v;                    // child: mm + b2 (weighted later)
                }
            }
        }
    }
}

// ---------------- router: logits = h @ rw^T + rb ; softmax -> wbuf ----------------
__global__ void router_kernel(const float* __restrict__ h, const float* __restrict__ rw,
                              const float* __restrict__ rb, float* __restrict__ wout,
                              const int* __restrict__ idxslot, int off)
{
    int row  = blockIdx.x;
    int gidx = off + idxslot[0];
    const float* hp  = h  + (size_t)row * HDIM;
    const float* rwp = rw + (size_t)gidx * CNUM * HDIM;
    float p0 = 0, p1 = 0, p2 = 0, p3 = 0;
    for (int i = threadIdx.x; i < HDIM; i += 256) {
        float v = hp[i];
        p0 += v * rwp[i];
        p1 += v * rwp[HDIM + i];
        p2 += v * rwp[2 * HDIM + i];
        p3 += v * rwp[3 * HDIM + i];
    }
    __shared__ float sh[4 * 256];
    sh[threadIdx.x] = p0; sh[256 + threadIdx.x] = p1;
    sh[512 + threadIdx.x] = p2; sh[768 + threadIdx.x] = p3;
    __syncthreads();
    for (int st = 128; st > 0; st >>= 1) {
        if (threadIdx.x < st) {
            sh[threadIdx.x]       += sh[threadIdx.x + st];
            sh[256 + threadIdx.x] += sh[256 + threadIdx.x + st];
            sh[512 + threadIdx.x] += sh[512 + threadIdx.x + st];
            sh[768 + threadIdx.x] += sh[768 + threadIdx.x + st];
        }
        __syncthreads();
    }
    if (threadIdx.x == 0) {
        const float* rbp = rb + (size_t)gidx * CNUM;
        float l[4]; float mx = -1e30f;
        for (int c = 0; c < 4; c++) { l[c] = sh[c * 256] + rbp[c]; mx = fmaxf(mx, l[c]); }
        float sum = 0.f;
        for (int c = 0; c < 4; c++) { l[c] = expf(l[c] - mx); sum += l[c]; }
        for (int c = 0; c < 4; c++) wout[row * 4 + c] = l[c] / sum;
    }
}

// separate 1-thread kernel so routing-state update never races with readers
__global__ void route_update_kernel(const float* __restrict__ wout, int* __restrict__ idx,
                                    int child_off)
{
    float mx = wout[0]; int best = 0;
    for (int c = 1; c < 4; c++) { if (wout[c] > mx) { mx = wout[c]; best = c; } }
    int lo = idx[0];
    idx[1] = child_off + lo * 4;   // absolute gidx of first child
    idx[0] = lo * 4 + best;        // local index at next level
}

// ---------------- cur = 2*cur + sum_c w[b,c]*couts[c] ----------------
__global__ void combine_kernel(float* __restrict__ cur, const float* __restrict__ couts,
                               const float* __restrict__ wbuf, int n)
{
    int i = blockIdx.x * blockDim.x + threadIdx.x;
    if (i >= n) return;
    int b = i / DDIM;
    float v = 2.f * cur[i];
    #pragma unroll
    for (int c = 0; c < 4; c++)
        v += wbuf[b * 4 + c] * couts[(size_t)c * ((size_t)BDIM * DDIM) + i];
    cur[i] = v;
}

extern "C" void kernel_launch(void* const* d_in, const int* in_sizes, int n_in,
                              void* d_out, int out_size, void* d_ws, size_t ws_size,
                              hipStream_t stream)
{
    (void)in_sizes; (void)n_in; (void)out_size; (void)ws_size;
    const float* x     = (const float*)d_in[0];
    const float* ln1_w = (const float*)d_in[1];
    const float* ln1_b = (const float*)d_in[2];
    const float* fc1_w = (const float*)d_in[3];
    const float* fc1_b = (const float*)d_in[4];
    const float* ln2_w = (const float*)d_in[5];
    const float* ln2_b = (const float*)d_in[6];
    const float* fc2_w = (const float*)d_in[7];
    const float* fc2_b = (const float*)d_in[8];
    const float* rw    = (const float*)d_in[9];
    const float* rb    = (const float*)d_in[10];

    char* ws = (char*)d_ws;
    float*    cur   = (float*)(ws);                        // 1 MB
    _Float16* xn16  = (_Float16*)(ws + (1u  << 20));       // 4 x 0.5 MB
    float*    hbuf  = (float*)(ws + (3u  << 20));          // 4 x 4 MB
    _Float16* hn16  = (_Float16*)(ws + (19u << 20));       // 4 x 2 MB
    float*    couts = (float*)(ws + (27u << 20));          // 4 x 1 MB
    _Float16* w16a  = (_Float16*)(ws + (31u << 20));       // 4 x 2 MB (fc1 f16)
    _Float16* w16b  = (_Float16*)(ws + (39u << 20));       // 4 x 2 MB (fc2 f16)
    float*    wbuf  = (float*)(ws + (47u << 20));          // 8 KB
    int*      idx   = (int*)(ws + (47u << 20) + 8192);     // [local, child_base]

    const size_t BD = (size_t)BDIM * DDIM;
    const size_t BH = (size_t)BDIM * HDIM;
    const int WE = HDIM * DDIM;                            // weight elems per agent (both fc)
    const int OFFS[5] = {0, 1, 5, 21, 85};

    init_kernel<<<dim3((unsigned)((BD + 255) / 256)), 256, 0, stream>>>(x, cur, idx, (int)BD);

    for (int d = 0; d <= 3; d++) {
        // ---- parent agent (gidx = OFFS[d] + idx[0]) ----
        ln_kernel<<<dim3(BDIM, 1), 256, 0, stream>>>(cur, xn16, ln1_w, ln1_b,
                                                     idx, OFFS[d], DDIM, 0, BD);
        convw_kernel<<<dim3(WE / 1024, 1, 1), 256, 0, stream>>>(fc1_w, w16a, idx, OFFS[d], WE);
        wmma_gemm<<<dim3((BDIM / 256) * (HDIM / 64), 1, 1), 128, 0, stream>>>(
            xn16, w16a, fc1_b, hbuf, idx, OFFS[d], BDIM, HDIM, DDIM, BD, BH, EPI_GELU);
        ln_kernel<<<dim3(BDIM, 1), 256, 0, stream>>>(hbuf, hn16, ln2_w, ln2_b,
                                                     idx, OFFS[d], HDIM, 0, BH);
        convw_kernel<<<dim3(WE / 1024, 1, 1), 256, 0, stream>>>(fc2_w, w16b, idx, OFFS[d], WE);
        wmma_gemm<<<dim3((BDIM / 256) * (DDIM / 64), 1, 1), 128, 0, stream>>>(
            hn16, w16b, fc2_b, cur, idx, OFFS[d], BDIM, DDIM, HDIM, BH, 0, EPI_RESID);

        if (d < 3) {
            // router on parent's pre-update h; then update routing state
            router_kernel<<<dim3(BDIM), 256, 0, stream>>>(hbuf, rw, rb, wbuf, idx, OFFS[d]);
            route_update_kernel<<<1, 1, 0, stream>>>(wbuf, idx, OFFS[d + 1]);

            // ---- 4 children batched in grid.z (gidx = idx[1] + z) ----
            ln_kernel<<<dim3(BDIM, 4), 256, 0, stream>>>(cur, xn16, ln1_w, ln1_b,
                                                         idx + 1, 0, DDIM, 0, BD);
            convw_kernel<<<dim3(WE / 1024, 1, 4), 256, 0, stream>>>(fc1_w, w16a, idx + 1, 0, WE);
            wmma_gemm<<<dim3((BDIM / 256) * (HDIM / 64), 1, 4), 128, 0, stream>>>(
                xn16, w16a, fc1_b, hbuf, idx + 1, 0, BDIM, HDIM, DDIM, BD, BH, EPI_GELU);
            ln_kernel<<<dim3(BDIM, 4), 256, 0, stream>>>(hbuf, hn16, ln2_w, ln2_b,
                                                         idx + 1, 0, HDIM, BH, BH);
            convw_kernel<<<dim3(WE / 1024, 1, 4), 256, 0, stream>>>(fc2_w, w16b, idx + 1, 0, WE);
            wmma_gemm<<<dim3((BDIM / 256) * (DDIM / 64), 1, 4), 128, 0, stream>>>(
                hn16, w16b, fc2_b, couts, idx + 1, 0, BDIM, DDIM, HDIM, BH, BD, EPI_STORE);
            combine_kernel<<<dim3((unsigned)((BD + 255) / 256)), 256, 0, stream>>>(
                cur, couts, wbuf, (int)BD);
        }
    }
    hipMemcpyAsync(d_out, cur, BD * sizeof(float), hipMemcpyDeviceToDevice, stream);
}
